// DCGRUCell_4071628996928
// MI455X (gfx1250) — matrix operand order repaired and checked by reading, not compiled, not run on hardware
//
#include <hip/hip_runtime.h>

// ---------------------------------------------------------------------------
// DCGRU cell for MI455X (gfx1250, wave32, WMMA).
// B=16, N=8192, D_IN=D_H=64, D_CAT=128, NUM_MAT=5, K_TOT=640.
// Dominant work: 8 GEMMs  S(8192x8192) @ X(8192x2048) -> bf16.
// Supports pre-converted to bf16 (one 128MB support fits the 192MB L2, and
// per-GEMM HBM traffic halves to 128MB).  GEMM A tiles are moved with
// global_load_async_to_lds_b128 (ASYNCcnt) into a 3-buffer LDS rotation so
// the HBM stream never passes through VGPRs; B tiles are pipelined 2 tiles
// ahead in registers.  The per-step barrier waits ONLY dscnt+asynccnt (a
// hand-rolled split barrier), so the B prefetch stays in flight across it.
// ---------------------------------------------------------------------------

typedef __attribute__((ext_vector_type(8)))  float  v8f;
typedef __attribute__((ext_vector_type(16))) __bf16 v16bf;
typedef __attribute__((ext_vector_type(8)))  __bf16 v8bf;
typedef __attribute__((ext_vector_type(4)))  __bf16 v4bf;

#define NN      8192            // nodes
#define NC      2048            // B * D_CAT packed columns
#define BATCH   16
#define DH      64
#define DCAT    128
#define KTOT    640             // D_CAT * NUM_MAT
#define HALF_OUT (BATCH * NN * DH)   // 8388608

__device__ __forceinline__ v8f wmma_bf16(v16bf a, v16bf b, v8f c) {
  // v_wmma_f32_16x16x32_bf16 : D = A(16x32) * B(32x16) + C(16x16 f32)
  return __builtin_amdgcn_wmma_f32_16x16x32_bf16(
      false, a, false, b, (short)0, c, false, false);
}

__device__ __forceinline__ v16bf cat8(v8bf lo, v8bf hi) {
  return __builtin_shufflevector(lo, hi, 0,1,2,3,4,5,6,7,8,9,10,11,12,13,14,15);
}

// Async global->LDS copy, 16 bytes per lane.  Generic pointers to __shared__
// carry the LDS byte offset in their low 32 bits (aperture in the high bits),
// which is exactly what the VDST operand of the async op wants.
__device__ __forceinline__ void async_b128(const void* g, void* ldsp) {
  unsigned lds_off = (unsigned)(uintptr_t)ldsp;
  unsigned long long ga = (unsigned long long)(uintptr_t)g;
  asm volatile("global_load_async_to_lds_b128 %0, %1, off"
               :: "v"(lds_off), "v"(ga) : "memory");
}
__device__ __forceinline__ void wait_async(int n) {
  if (n == 0) asm volatile("s_wait_asynccnt 0x0" ::: "memory");
  else        asm volatile("s_wait_asynccnt 0x2" ::: "memory");
}
// Workgroup barrier that waits only DScnt (LDS stores visible) — NOT loadcnt.
// Nothing in the GEMM communicates through global memory between waves, so
// the conservative loadcnt wait __syncthreads() emits would only serialize
// the in-flight B prefetch.
__device__ __forceinline__ void block_sync_ds() {
  asm volatile(
      "s_wait_dscnt 0x0\n\t"
      "s_barrier_signal -1\n\t"
      "s_barrier_wait -1"
      ::: "memory");
}

// ---------------------------------------------------------------------------
// Elementwise support conversion: f32 -> bf16, both supports (2 x 8192^2).
// ---------------------------------------------------------------------------
__global__ __launch_bounds__(256) void cvt_support_kernel(
    const float* __restrict__ S, __bf16* __restrict__ Sb)
{
  size_t i4 = (size_t)blockIdx.x * 256 + threadIdx.x;   // float4 index
  const float4 v = *(const float4*)(S + i4 * 4);
  v4bf p;
  p[0] = (__bf16)v.x; p[1] = (__bf16)v.y;
  p[2] = (__bf16)v.z; p[3] = (__bf16)v.w;
  *(v4bf*)(Sb + i4 * 4) = p;
}

// ---------------------------------------------------------------------------
// Pack: X0[n][b*128+d] = bf16( d<64 ? inputs[b,n,d] : states[b,n,d-64] )
// ---------------------------------------------------------------------------
__global__ __launch_bounds__(256) void pack_x0_kernel(
    const float* __restrict__ inputs, const float* __restrict__ states,
    __bf16* __restrict__ X0)
{
  size_t idx = (size_t)blockIdx.x * 256 + threadIdx.x;   // [0, NN*NC)
  int n = (int)(idx >> 11);
  int c = (int)(idx & 2047);
  int b = c >> 7, d = c & 127;
  float v = (d < DH) ? inputs[((size_t)b * NN + n) * DH + d]
                     : states[((size_t)b * NN + n) * DH + (d - DH)];
  X0[idx] = (__bf16)v;
}

// Fill the "inputs" half of XC (the r*states half comes from proj_ru).
__global__ __launch_bounds__(256) void pack_xc_in_kernel(
    const float* __restrict__ inputs, __bf16* __restrict__ XC)
{
  size_t idx = (size_t)blockIdx.x * 256 + threadIdx.x;   // [0, NN*BATCH*DH)
  int n = (int)(idx >> 10);
  int b = (int)((idx >> 6) & 15);
  int d = (int)(idx & 63);
  XC[(size_t)n * NC + b * DCAT + d] = (__bf16)inputs[((size_t)b * NN + n) * DH + d];
}

// Transposed bf16 weights: WruT[o][f] (128x640), WcT[o][f] (64x640)
__global__ __launch_bounds__(256) void cvt_weights_kernel(
    const float* __restrict__ Wru, const float* __restrict__ Wc,
    __bf16* __restrict__ WruT, __bf16* __restrict__ WcT)
{
  int idx = blockIdx.x * 256 + threadIdx.x;
  if (idx < KTOT * (2 * DH)) {                 // 640*128
    int o = idx / KTOT, f = idx % KTOT;
    WruT[idx] = (__bf16)Wru[(size_t)f * (2 * DH) + o];
  } else {
    int j = idx - KTOT * (2 * DH);
    if (j < KTOT * DH) {
      int o = j / KTOT, f = j % KTOT;
      WcT[j] = (__bf16)Wc[(size_t)f * DH + o];
    }
  }
}

// ---------------------------------------------------------------------------
// Big GEMM: Y(8192x2048,bf16) = S(8192x8192,bf16) @ X(8192x2048,bf16)
// mode==1 Chebyshev fuse: Y = 2*S@X - Cb
// Block: 256 threads (8 waves), tile 128x128, BK=32, 3-buffer LDS pipeline.
// Wave grid 2x4; each wave: 64 rows x 32 cols = 4x2 WMMA accum tiles.
// ---------------------------------------------------------------------------
#define AS 40        // LDS A row stride (bf16 elems), padded
#define BS 40        // LDS B(transposed) row stride
#define NKS (NN / 32)

__global__ __launch_bounds__(256) void gemm_cheb_kernel(
    const __bf16* __restrict__ S, const __bf16* __restrict__ X,
    __bf16* __restrict__ Y, const __bf16* __restrict__ Cb, int mode)
{
  __shared__ __bf16 Alds[3][128 * AS];   // [buf][row][k]
  __shared__ __bf16 Blds[3][128 * BS];   // [buf][col][k]  (transposed)

  const int tid = threadIdx.x;
  const int wid = tid >> 5;
  const int lane = tid & 31;
  const int ln = lane & 15;          // fragment row/col within 16
  const int lg = lane >> 4;          // half-wave group (0/1)
  const int waveRow = wid >> 2;      // 0..1 -> 64-row band
  const int waveCol = wid & 3;       // 0..3 -> 32-col band
  const int rowBase = blockIdx.y * 128;
  const int colBase = blockIdx.x * 128;

  // ---- A staging: 2 async 16B chunks per thread (128x32 bf16 tile) -------
  const __bf16* aG[2];
  int aOff[2];
  #pragma unroll
  for (int i = 0; i < 2; ++i) {        // 512 chunks / 256 threads
    int s = tid + 256 * i;
    int row = s >> 2, k8 = (s & 3) * 8;
    aG[i]  = S + (size_t)(rowBase + row) * NN + k8;
    aOff[i] = row * AS + k8;
  }
  // ---- B staging: one 32B (16 bf16) chunk per thread ---------------------
  const int bKrow = tid >> 3;          // k row 0..31
  const int bCb   = (tid & 7) * 16;    // col base 0..112
  const __bf16* bG = X + (size_t)bKrow * NC + colBase + bCb;

  v8f acc[4][2];
  const v8f vzero = {0.f, 0.f, 0.f, 0.f, 0.f, 0.f, 0.f, 0.f};
  #pragma unroll
  for (int i = 0; i < 4; ++i)
    #pragma unroll
    for (int j = 0; j < 2; ++j) acc[i][j] = vzero;

  // ---- prologue: tiles 0 and 1 -------------------------------------------
  #pragma unroll
  for (int i = 0; i < 2; ++i) {
    async_b128(aG[i],      &Alds[0][aOff[i]]);   // tile 0
    async_b128(aG[i] + 32, &Alds[1][aOff[i]]);   // tile 1
  }
  {
    v16bf b0 = *(const v16bf*)(bG);              // tile 0
    #pragma unroll
    for (int j = 0; j < 16; ++j) Blds[0][(bCb + j) * BS + bKrow] = b0[j];
  }
  v16bf bCur = *(const v16bf*)(bG + (size_t)32 * NC);            // tile 1
  wait_async(0);
  block_sync_ds();

  // ---- main loop: async A two tiles ahead, B regs two tiles ahead --------
  int cur = 0;
  for (int it = 0; it < NKS; ++it) {
    int nb1 = cur + 1; if (nb1 == 3) nb1 = 0;
    int nb2 = nb1 + 1; if (nb2 == 3) nb2 = 0;

    if (it + 2 < NKS) {
      #pragma unroll
      for (int i = 0; i < 2; ++i)
        async_b128(aG[i] + (size_t)(it + 2) * 32, &Alds[nb2][aOff[i]]);
    }
    if (it + 1 < NKS) {
      #pragma unroll
      for (int j = 0; j < 16; ++j) Blds[nb1][(bCb + j) * BS + bKrow] = bCur[j];
    }
    if (it + 2 < NKS)
      bCur = *(const v16bf*)(bG + (size_t)(it + 2) * 32 * NC);

    // compute tile it from LDS buffer `cur`
    v16bf afrag[4];
    #pragma unroll
    for (int rt = 0; rt < 4; ++rt) {
      int m = waveRow * 64 + rt * 16 + ln;
      v8bf lo = *(const v8bf*)&Alds[cur][m * AS + lg * 8];
      v8bf hi = *(const v8bf*)&Alds[cur][m * AS + 16 + lg * 8];
      afrag[rt] = cat8(lo, hi);
    }
    v16bf bfrag[2];
    #pragma unroll
    for (int ct = 0; ct < 2; ++ct) {
      int c = waveCol * 32 + ct * 16 + ln;
      v8bf lo = *(const v8bf*)&Blds[cur][c * BS + lg * 8];
      v8bf hi = *(const v8bf*)&Blds[cur][c * BS + 16 + lg * 8];
      bfrag[ct] = cat8(lo, hi);
    }
    #pragma unroll
    for (int rt = 0; rt < 4; ++rt)
      #pragma unroll
      for (int ct = 0; ct < 2; ++ct)
        acc[rt][ct] = wmma_bf16(afrag[rt], bfrag[ct], acc[rt][ct]);

    if (it + 1 < NKS) {
      wait_async(2);        // tile it+1 copies done; it+2 may stay in flight
      block_sync_ds();      // barrier w/o loadcnt: B prefetch stays in flight
    }
    cur = nb1;
  }

  // ---- epilogue: (optional Chebyshev) + bf16 store -----------------------
  #pragma unroll
  for (int rt = 0; rt < 4; ++rt) {
    #pragma unroll
    for (int ct = 0; ct < 2; ++ct) {
      int gc = colBase + waveCol * 32 + ct * 16 + ln;
      #pragma unroll
      for (int r = 0; r < 8; ++r) {
        int gm = rowBase + waveRow * 64 + rt * 16 + r + 8 * lg;
        float v = acc[rt][ct][r];
        if (mode == 1) v = 2.0f * v - (float)Cb[(size_t)gm * NC + gc];
        Y[(size_t)gm * NC + gc] = (__bf16)v;
      }
    }
  }
}

// ---------------------------------------------------------------------------
// Projection RU:  ru = sigmoid( h @ W_ru + b_ru ),  h = [F0|F1|F2|F3|F4]
// Each wave: one 16-row block (one b, 16 n) x 128 outputs (8 WMMA col tiles).
// Epilogue: r -> XC r*states half;  u -> u_ws (f32).
// ---------------------------------------------------------------------------
__global__ __launch_bounds__(256) void proj_ru_kernel(
    const __bf16* __restrict__ F0, const __bf16* __restrict__ F1,
    const __bf16* __restrict__ F2, const __bf16* __restrict__ F3,
    const __bf16* __restrict__ F4, const __bf16* __restrict__ WT,
    const float* __restrict__ bias, const float* __restrict__ states,
    float* __restrict__ u_ws, __bf16* __restrict__ XC)
{
  const int tid = threadIdx.x;
  const int wid = tid >> 5, lane = tid & 31;
  const int ln = lane & 15, lg = lane >> 4;
  const int rowBlock = blockIdx.x * 8 + wid;
  const int rb = rowBlock * 16;
  const int b = rb >> 13;          // NN = 8192 rows per batch
  const int n0 = rb & (NN - 1);

  const __bf16* feats[5] = {F0, F1, F2, F3, F4};

  v8f acc[8];
  const v8f vzero = {0.f, 0.f, 0.f, 0.f, 0.f, 0.f, 0.f, 0.f};
  #pragma unroll
  for (int i = 0; i < 8; ++i) acc[i] = vzero;

  #pragma unroll
  for (int fk = 0; fk < 5; ++fk) {
    const __bf16* F = feats[fk];
    #pragma unroll
    for (int ks = 0; ks < 4; ++ks) {
      int d0 = ks * 32;
      const __bf16* ap = F + (size_t)(n0 + ln) * NC + b * DCAT + d0 + lg * 8;
      v16bf a = cat8(*(const v8bf*)ap, *(const v8bf*)(ap + 16));
      int f0 = fk * DCAT + d0;
      #pragma unroll
      for (int ot = 0; ot < 8; ++ot) {
        int o = ot * 16 + ln;
        const __bf16* bp = WT + (size_t)o * KTOT + f0 + lg * 8;
        v16bf bb = cat8(*(const v8bf*)bp, *(const v8bf*)(bp + 16));
        acc[ot] = wmma_bf16(a, bb, acc[ot]);
      }
    }
  }

  #pragma unroll
  for (int ot = 0; ot < 8; ++ot) {
    int o = ot * 16 + ln;
    float bo = bias[o];
    #pragma unroll
    for (int r = 0; r < 8; ++r) {
      int n = n0 + r + 8 * lg;
      float val = acc[ot][r] + bo;
      float sg = 1.0f / (1.0f + __expf(-val));
      size_t bn = (size_t)b * NN + n;
      if (o < DH) {               // r gate -> xc = [inputs, r*states]
        float st = states[bn * DH + o];
        XC[(size_t)n * NC + b * DCAT + DH + o] = (__bf16)(sg * st);
      } else {                    // u gate
        u_ws[bn * DH + (o - DH)] = sg;
      }
    }
  }
}

// ---------------------------------------------------------------------------
// Projection C + GRU combine: c = tanh(h_c @ W_c + b_c);
// new = u*states + (1-u)*c ; written to both tuple halves of d_out.
// ---------------------------------------------------------------------------
__global__ __launch_bounds__(256) void proj_c_final_kernel(
    const __bf16* __restrict__ F0, const __bf16* __restrict__ F1,
    const __bf16* __restrict__ F2, const __bf16* __restrict__ F3,
    const __bf16* __restrict__ F4, const __bf16* __restrict__ WT,
    const float* __restrict__ bias, const float* __restrict__ u_ws,
    const float* __restrict__ states, float* __restrict__ out)
{
  const int tid = threadIdx.x;
  const int wid = tid >> 5, lane = tid & 31;
  const int ln = lane & 15, lg = lane >> 4;
  const int rowBlock = blockIdx.x * 8 + wid;
  const int rb = rowBlock * 16;
  const int b = rb >> 13;
  const int n0 = rb & (NN - 1);

  const __bf16* feats[5] = {F0, F1, F2, F3, F4};

  v8f acc[4];
  const v8f vzero = {0.f, 0.f, 0.f, 0.f, 0.f, 0.f, 0.f, 0.f};
  #pragma unroll
  for (int i = 0; i < 4; ++i) acc[i] = vzero;

  #pragma unroll
  for (int fk = 0; fk < 5; ++fk) {
    const __bf16* F = feats[fk];
    #pragma unroll
    for (int ks = 0; ks < 4; ++ks) {
      int d0 = ks * 32;
      const __bf16* ap = F + (size_t)(n0 + ln) * NC + b * DCAT + d0 + lg * 8;
      v16bf a = cat8(*(const v8bf*)ap, *(const v8bf*)(ap + 16));
      int f0 = fk * DCAT + d0;
      #pragma unroll
      for (int ot = 0; ot < 4; ++ot) {
        int o = ot * 16 + ln;
        const __bf16* bp = WT + (size_t)o * KTOT + f0 + lg * 8;
        v16bf bb = cat8(*(const v8bf*)bp, *(const v8bf*)(bp + 16));
        acc[ot] = wmma_bf16(a, bb, acc[ot]);
      }
    }
  }

  #pragma unroll
  for (int ot = 0; ot < 4; ++ot) {
    int o = ot * 16 + ln;
    float bo = bias[o];
    #pragma unroll
    for (int r = 0; r < 8; ++r) {
      int n = n0 + r + 8 * lg;
      float c = tanhf(acc[ot][r] + bo);
      size_t idx = ((size_t)b * NN + n) * DH + o;
      float u = u_ws[idx];
      float nv = u * states[idx] + (1.0f - u) * c;
      out[idx] = nv;
      out[HALF_OUT + idx] = nv;
    }
  }
}

// ---------------------------------------------------------------------------
// Host launch
// ---------------------------------------------------------------------------
extern "C" void kernel_launch(void* const* d_in, const int* in_sizes, int n_in,
                              void* d_out, int out_size, void* d_ws, size_t ws_size,
                              hipStream_t stream) {
  (void)in_sizes; (void)n_in; (void)out_size; (void)ws_size;

  const float* inputs  = (const float*)d_in[0];   // (16, 8192, 64)
  const float* states  = (const float*)d_in[1];   // (16, 8192, 64)
  const float* sup     = (const float*)d_in[2];   // (2, 8192, 8192)
  const float* W_ru    = (const float*)d_in[3];   // (640, 128)
  const float* b_ru    = (const float*)d_in[4];   // (128,)
  const float* W_c     = (const float*)d_in[5];   // (640, 64)
  const float* b_c     = (const float*)d_in[6];   // (64,)
  float* out = (float*)d_out;

  // Workspace carve-up (~496 MB)
  const size_t featBytes = (size_t)NN * NC * sizeof(__bf16);   // 32 MB
  char* w = (char*)d_ws;
  __bf16* X0   = (__bf16*)(w);
  __bf16* F1   = (__bf16*)(w + 1 * featBytes);
  __bf16* F2   = (__bf16*)(w + 2 * featBytes);
  __bf16* F3   = (__bf16*)(w + 3 * featBytes);
  __bf16* F4   = (__bf16*)(w + 4 * featBytes);
  __bf16* XC   = (__bf16*)(w + 5 * featBytes);
  float*  u_ws = (float*)(w + 6 * featBytes);                  // 32 MB f32
  __bf16* WruT = (__bf16*)(w + 6 * featBytes + (size_t)BATCH * NN * DH * sizeof(float));
  __bf16* WcT  = WruT + (size_t)(2 * DH) * KTOT;
  __bf16* S0b  = (__bf16*)(w + 251658240);                     // 240 MB offset
  __bf16* S1b  = S0b + (size_t)NN * NN;

  const dim3 blk(256);
  const dim3 gemmGrid(NC / 128, NN / 128);   // 16 x 64

  // Conversions / packing
  cvt_support_kernel<<<(unsigned)((size_t)2 * NN * NN / 4 / 256), blk, 0, stream>>>(sup, S0b);
  pack_x0_kernel<<<(unsigned)((size_t)NN * NC / 256), blk, 0, stream>>>(inputs, states, X0);
  pack_xc_in_kernel<<<(unsigned)((size_t)NN * BATCH * DH / 256), blk, 0, stream>>>(inputs, XC);
  cvt_weights_kernel<<<(KTOT * (2 * DH) + KTOT * DH + 255) / 256, blk, 0, stream>>>(
      W_ru, W_c, WruT, WcT);

  // gconv #1 diffusion GEMMs
  gemm_cheb_kernel<<<gemmGrid, blk, 0, stream>>>(S0b, X0, F1, X0, 0);  // f1 = s0@x
  gemm_cheb_kernel<<<gemmGrid, blk, 0, stream>>>(S0b, F1, F2, X0, 1);  // f2 = 2*s0@f1 - x
  gemm_cheb_kernel<<<gemmGrid, blk, 0, stream>>>(S1b, X0, F3, X0, 0);  // f3 = s1@x
  gemm_cheb_kernel<<<gemmGrid, blk, 0, stream>>>(S1b, F3, F4, X0, 1);  // f4 = 2*s1@f3 - x

  // ru projection + sigmoid; builds XC r*states half and u
  proj_ru_kernel<<<(BATCH * NN / 16) / 8, blk, 0, stream>>>(
      X0, F1, F2, F3, F4, WruT, b_ru, states, u_ws, XC);

  // gconv #2 diffusion GEMMs (reuse F buffers)
  gemm_cheb_kernel<<<gemmGrid, blk, 0, stream>>>(S0b, XC, F1, XC, 0);
  gemm_cheb_kernel<<<gemmGrid, blk, 0, stream>>>(S0b, F1, F2, XC, 1);
  gemm_cheb_kernel<<<gemmGrid, blk, 0, stream>>>(S1b, XC, F3, XC, 0);
  gemm_cheb_kernel<<<gemmGrid, blk, 0, stream>>>(S1b, F3, F4, XC, 1);

  // c projection + tanh + GRU combine -> both tuple outputs
  proj_c_final_kernel<<<(BATCH * NN / 16) / 8, blk, 0, stream>>>(
      XC, F1, F2, F3, F4, WcT, b_c, u_ws, states, out);
}
